// GATGCN_74655121539488
// MI455X (gfx1250) — matrix operand order repaired and checked by reading, not compiled, not run on hardware
//
#include <hip/hip_runtime.h>

// ---------------- problem constants ----------------
#define N_NODES 30000
#define E_RAW   300000
#define E_TOT   330000      // + self loops
#define NGRAPH  512
#define HEADS   10
#define CHC     78
#define HC      780         // HEADS*CHC
#define SEQ     1000
#define CONVO   32
#define CONVK   8
#define CONVP   121         // 128 - 8 + 1
#define EMB_V   26
#define EMB_D   128
#define CONV_KT 250         // 1000/4 K-tiles of 32

typedef __attribute__((ext_vector_type(16))) __bf16 v16bf;
typedef __attribute__((ext_vector_type(8)))  __bf16 v8bf;
typedef __attribute__((ext_vector_type(8)))  float  v8f;

__device__ __forceinline__ unsigned short f2bfu(float f) {
  unsigned u = __builtin_bit_cast(unsigned, f);
  return (unsigned short)((u + 0x7FFFu + ((u >> 16) & 1u)) >> 16);   // RNE
}
__device__ __forceinline__ __bf16 f2bf(float f) {
  unsigned short s = f2bfu(f);
  return __builtin_bit_cast(__bf16, s);
}

__device__ __forceinline__ void atomicMaxF(float* a, float v) {
  if (v >= 0.f) atomicMax((int*)a, __float_as_int(v));
  else          atomicMin((unsigned int*)a, __float_as_uint(v));
}

__device__ __forceinline__ void edge_sd(const int* __restrict__ ei, int e, int& s, int& d) {
  if (e < E_RAW) { s = ei[e]; d = ei[E_RAW + e]; }
  else           { s = e - E_RAW; d = s; }               // self loop
}

// A fragment: two contiguous 8-bf16 runs (K = +0..7 and +16..23 relative to lane base)
__device__ __forceinline__ v16bf ld_afrag(const __bf16* p) {
  const v8bf* pa = (const v8bf*)p;
  v8bf lo = pa[0];
  v8bf hi = pa[2];
  return __builtin_shufflevector(lo, hi, 0,1,2,3,4,5,6,7,8,9,10,11,12,13,14,15);
}

// ---------------- utility fill ----------------
__global__ void k_fill(float* p, float v, long n) {
  long i = (long)blockIdx.x * blockDim.x + threadIdx.x;
  if (i < n) p[i] = v;
}

// ---------------- operand packing ----------------
// A -> bf16 row-major, rows padded to Mt*64, cols padded to Kp (mult of 32)
__global__ void k_packA(const float* __restrict__ A, __bf16* __restrict__ Ap,
                        int M, int K, int Kp, long total) {
  long i = (long)blockIdx.x * blockDim.x + threadIdx.x;
  if (i >= total) return;
  int m = (int)(i / Kp), k = (int)(i - (long)m * Kp);
  Ap[i] = (m < M && k < K) ? f2bf(A[(long)m * K + k]) : f2bf(0.f);
}

// B -> fragment-major: [nt][kt][lane][e] ; value = B[kt*32 + 16*(e>>3) + 8*(lane>>4) + (e&7), nt*16 + (lane&15)]
__global__ void k_packB(const float* __restrict__ B, __bf16* __restrict__ Bp,
                        int K, int N, int Kt, int Nt) {
  long total = (long)Kt * Nt * 512;
  long i = (long)blockIdx.x * blockDim.x + threadIdx.x;
  if (i >= total) return;
  int e    = (int)(i & 15);
  int lane = (int)((i >> 4) & 31);
  long tile = i >> 9;
  int kt = (int)(tile % Kt);
  int nt = (int)(tile / Kt);
  int kk = (kt << 5) + ((e & 8) << 1) + ((lane >> 4) << 3) + (e & 7);
  int nn = (nt << 4) + (lane & 15);
  Bp[i] = (kk < K && nn < N) ? f2bf(B[(long)kk * N + nn]) : f2bf(0.f);
}

// W_conv[32,1000,8] -> fragment-major [nt(2)][kt(250)][lane][e]; "B"[kk,o] = Wc[o, kk>>3, kk&7]
__global__ void k_packWc(const float* __restrict__ Wc, __bf16* __restrict__ Bp) {
  long total = (long)2 * CONV_KT * 512;
  long i = (long)blockIdx.x * blockDim.x + threadIdx.x;
  if (i >= total) return;
  int e    = (int)(i & 15);
  int lane = (int)((i >> 4) & 31);
  long tile = i >> 9;
  int kt = (int)(tile % CONV_KT);
  int nt = (int)(tile / CONV_KT);
  int kk = (kt << 5) + ((e & 8) << 1) + ((lane >> 4) << 3) + (e & 7);
  int o  = (nt << 4) + (lane & 15);
  Bp[i] = f2bf(Wc[(long)o * (SEQ * CONVK) + kk]);
}

// ---------------- packed bf16 WMMA GEMM: C = act(A@B + bias) ----------------
// Each wave: 64x32 C tile (4 m-tiles x 2 n-tiles), software-pipelined K loop.
// Pack buffers have >=1 tile of slack so the pipeline's one-tile over-read is safe.
__global__ void __launch_bounds__(256)
k_gemm_packed(const __bf16* __restrict__ Ap, const __bf16* __restrict__ Bp,
              const float* __restrict__ bias, float* __restrict__ C,
              int M, int N, int Kp, int ldc, int relu)
{
  int wave  = blockIdx.x * (blockDim.x >> 5) + (threadIdx.x >> 5);
  int lane  = threadIdx.x & 31;
  int lhalf = lane >> 4;
  int lrow  = lane & 15;
  int Mt  = (M + 63) >> 6;
  int Nt2 = (N + 31) >> 5;                 // pairs of n-tiles
  if (wave >= Mt * Nt2) return;
  int mt  = wave / Nt2;
  int nt2 = wave - mt * Nt2;
  int mbase = mt << 6;
  int nbase = nt2 << 5;
  int Kt = Kp >> 5;

  long bstride = ((long)Kt << 9);          // elems between adjacent n-tiles
  const __bf16* bptr = Bp + (long)(nt2 * 2) * bstride + (lane << 4);
  const __bf16* aptr = Ap + (long)(mbase + lrow) * Kp + (lhalf << 3);
  long arow = (long)16 * Kp;               // stride between m-tiles

  v8f acc[4][2] = {};
  v16bf bc0 = *(const v16bf*)bptr;
  v16bf bc1 = *(const v16bf*)(bptr + bstride);
  v16bf a0 = ld_afrag(aptr);
  v16bf a1 = ld_afrag(aptr + arow);
  v16bf a2 = ld_afrag(aptr + 2 * arow);
  v16bf a3 = ld_afrag(aptr + 3 * arow);

  for (int kt = 0; kt < Kt; ++kt) {
    bptr += 512;
    aptr += 32;
    v16bf bn0 = *(const v16bf*)bptr;                 // prefetch next K-step
    v16bf bn1 = *(const v16bf*)(bptr + bstride);
    v16bf n0 = ld_afrag(aptr);
    v16bf n1 = ld_afrag(aptr + arow);
    v16bf n2 = ld_afrag(aptr + 2 * arow);
    v16bf n3 = ld_afrag(aptr + 3 * arow);

    acc[0][0] = __builtin_amdgcn_wmma_f32_16x16x32_bf16(false, a0, false, bc0, (short)0, acc[0][0], false, false);
    acc[0][1] = __builtin_amdgcn_wmma_f32_16x16x32_bf16(false, a0, false, bc1, (short)0, acc[0][1], false, false);
    acc[1][0] = __builtin_amdgcn_wmma_f32_16x16x32_bf16(false, a1, false, bc0, (short)0, acc[1][0], false, false);
    acc[1][1] = __builtin_amdgcn_wmma_f32_16x16x32_bf16(false, a1, false, bc1, (short)0, acc[1][1], false, false);
    acc[2][0] = __builtin_amdgcn_wmma_f32_16x16x32_bf16(false, a2, false, bc0, (short)0, acc[2][0], false, false);
    acc[2][1] = __builtin_amdgcn_wmma_f32_16x16x32_bf16(false, a2, false, bc1, (short)0, acc[2][1], false, false);
    acc[3][0] = __builtin_amdgcn_wmma_f32_16x16x32_bf16(false, a3, false, bc0, (short)0, acc[3][0], false, false);
    acc[3][1] = __builtin_amdgcn_wmma_f32_16x16x32_bf16(false, a3, false, bc1, (short)0, acc[3][1], false, false);

    bc0 = bn0; bc1 = bn1;
    a0 = n0; a1 = n1; a2 = n2; a3 = n3;
  }

#pragma unroll
  for (int u = 0; u < 2; ++u) {
    int nn = nbase + (u << 4) + lrow;
    if (nn < N) {
      float bv = bias ? bias[nn] : 0.f;
#pragma unroll
      for (int t = 0; t < 4; ++t) {
#pragma unroll
        for (int j = 0; j < 8; ++j) {
          int mm = mbase + (t << 4) + j + (lhalf << 3);
          if (mm < M) {
            float v = acc[t][u][j] + bv;
            if (relu) v = fmaxf(v, 0.f);
            C[(long)mm * ldc + nn] = v;
          }
        }
      }
    }
  }
}

// ---------------- GAT attention projections ----------------
__global__ void k_att(const float* __restrict__ h, const float* __restrict__ ws,
                      const float* __restrict__ wd, float* __restrict__ aS,
                      float* __restrict__ aD) {
  int i = blockIdx.x * blockDim.x + threadIdx.x;
  if (i >= N_NODES * HEADS) return;
  int n = i / HEADS, hd = i - n * HEADS;
  const float* hp = h + (long)n * HC + hd * CHC;
  const float* ps = ws + hd * CHC;
  const float* pd = wd + hd * CHC;
  float ss = 0.f, sd = 0.f;
  for (int c = 0; c < CHC; ++c) { float v = hp[c]; ss += v * ps[c]; sd += v * pd[c]; }
  aS[i] = ss; aD[i] = sd;
}

// ---------------- per-dst softmax over edges ----------------
__global__ void k_edge_max(const int* __restrict__ ei, const float* __restrict__ aS,
                           const float* __restrict__ aD, float* __restrict__ m) {
  long i = (long)blockIdx.x * blockDim.x + threadIdx.x;
  if (i >= (long)E_TOT * HEADS) return;
  int e = (int)(i / HEADS), hd = (int)(i - (long)e * HEADS);
  int s, d; edge_sd(ei, e, s, d);
  float l = aS[s * HEADS + hd] + aD[d * HEADS + hd];
  l = l > 0.f ? l : 0.2f * l;
  atomicMaxF(&m[d * HEADS + hd], l);
}

__global__ void k_edge_exp(const int* __restrict__ ei, const float* __restrict__ aS,
                           const float* __restrict__ aD, const float* __restrict__ m,
                           float* __restrict__ eWs, float* __restrict__ denom) {
  long i = (long)blockIdx.x * blockDim.x + threadIdx.x;
  if (i >= (long)E_TOT * HEADS) return;
  int e = (int)(i / HEADS), hd = (int)(i - (long)e * HEADS);
  int s, d; edge_sd(ei, e, s, d);
  float l = aS[s * HEADS + hd] + aD[d * HEADS + hd];
  l = l > 0.f ? l : 0.2f * l;
  float ev = __expf(l - m[d * HEADS + hd]);
  eWs[i] = ev;
  atomicAdd(&denom[d * HEADS + hd], ev);
}

__global__ void k_gat_aggr(const int* __restrict__ ei, const float* __restrict__ h,
                           const float* __restrict__ eWs, const float* __restrict__ denom,
                           float* __restrict__ xg) {
  long i = (long)blockIdx.x * blockDim.x + threadIdx.x;
  if (i >= (long)E_TOT * HEADS) return;
  int e = (int)(i / HEADS), hd = (int)(i - (long)e * HEADS);
  int s, d; edge_sd(ei, e, s, d);
  float alpha = eWs[i] / (denom[d * HEADS + hd] + 1e-16f);
  const float* hs = h + (long)s * HC + hd * CHC;
  float* xd = xg + (long)d * HC + hd * CHC;
  for (int c = 0; c < CHC; ++c) atomicAdd(&xd[c], hs[c] * alpha);
}

// ---------------- bias + relu in place ----------------
__global__ void k_bias_relu(float* __restrict__ x, const float* __restrict__ b,
                            long rows, int cols) {
  long i = (long)blockIdx.x * blockDim.x + threadIdx.x;
  if (i >= rows * cols) return;
  int c = (int)(i % cols);
  x[i] = fmaxf(x[i] + b[c], 0.f);
}

// ---------------- GCN degree / norm / aggregation ----------------
__global__ void k_deg(const int* __restrict__ ei, float* __restrict__ deg) {
  int e = blockIdx.x * blockDim.x + threadIdx.x;
  if (e >= E_TOT) return;
  int s, d; edge_sd(ei, e, s, d);
  atomicAdd(&deg[d], 1.0f);
}
__global__ void k_dinv(const float* __restrict__ deg, float* __restrict__ dinv) {
  int n = blockIdx.x * blockDim.x + threadIdx.x;
  if (n >= N_NODES) return;
  float dg = deg[n];
  dinv[n] = rsqrtf(dg < 1.f ? 1.f : dg);
}
__global__ void k_gcn_aggr(const int* __restrict__ ei, const float* __restrict__ xw,
                           const float* __restrict__ dinv, float* __restrict__ out) {
  long i = (long)blockIdx.x * blockDim.x + threadIdx.x;   // (edge, 78-ch chunk)
  if (i >= (long)E_TOT * HEADS) return;
  int e = (int)(i / HEADS), ch = (int)(i - (long)e * HEADS) * CHC;
  int s, d; edge_sd(ei, e, s, d);
  float nm = dinv[s] * dinv[d];
  const float* xs = xw + (long)s * HC + ch;
  float* od = out + (long)d * HC + ch;
  for (int c = 0; c < CHC; ++c) atomicAdd(&od[c], xs[c] * nm);
}

// ---------------- global max + mean pooling ----------------
__global__ void k_pool(const float* __restrict__ xg, const int* __restrict__ batch,
                       float* __restrict__ gmax, float* __restrict__ gsum) {
  long i = (long)blockIdx.x * blockDim.x + threadIdx.x;
  if (i >= (long)N_NODES * HC) return;
  int n = (int)(i / HC), c = (int)(i - (long)n * HC);
  int g = batch[n];
  float v = xg[i];
  atomicMaxF(&gmax[(long)g * HC + c], v);
  atomicAdd(&gsum[(long)g * HC + c], v);
}
__global__ void k_cnt(const int* __restrict__ batch, float* __restrict__ cnt) {
  int n = blockIdx.x * blockDim.x + threadIdx.x;
  if (n >= N_NODES) return;
  atomicAdd(&cnt[batch[n]], 1.f);
}
__global__ void k_pool_fin(const float* __restrict__ gmax, const float* __restrict__ gsum,
                           const float* __restrict__ cnt, float* __restrict__ xp) {
  long i = (long)blockIdx.x * blockDim.x + threadIdx.x;
  if (i >= (long)NGRAPH * HC) return;
  int g = (int)(i / HC), c = (int)(i - (long)g * HC);
  xp[(long)g * (2 * HC) + c]      = gmax[i];
  xp[(long)g * (2 * HC) + HC + c] = gsum[i] / cnt[g];
}

// ---------------- conv1d branch as WMMA implicit GEMM (one workgroup per graph) ----
// out[g,o,p] = sum_{i,k} emb[target[g,i], p+k] * W_conv[o,i,k] + b_conv[o]
__global__ void __launch_bounds__(256)
k_conv(const int* __restrict__ target, const float* __restrict__ emb,
       const __bf16* __restrict__ Wp, const float* __restrict__ bc,
       float* __restrict__ out) {
  __shared__ unsigned short semb[EMB_V * EMB_D + 8];   // +8 zero pad: no per-element guard
  __shared__ int stid[SEQ];
  int g = blockIdx.x, tid = threadIdx.x;
  for (int i = tid; i < EMB_V * EMB_D + 8; i += 256)
    semb[i] = (i < EMB_V * EMB_D) ? f2bfu(emb[i]) : (unsigned short)0;
  for (int i = tid; i < SEQ; i += 256) {
    int t = target[g * SEQ + i];
    stid[i] = t < 0 ? 0 : (t >= EMB_V ? EMB_V - 1 : t);
  }
  __syncthreads();

  int wave = tid >> 5, lane = tid & 31, lhalf = lane >> 4, lrow = lane & 15;
  int p = (wave << 4) + lrow;                      // 8 waves cover p = 0..127
  v8f acc0 = {0.f}, acc1 = {0.f};

  const __bf16* b0 = Wp + (lane << 4);
  const __bf16* b1 = Wp + (((long)CONV_KT) << 9) + (lane << 4);
  v16bf b0c = *(const v16bf*)b0;
  v16bf b1c = *(const v16bf*)b1;

  for (int i0 = 0; i0 < SEQ; i0 += 4) {            // K-step = 4 seq positions * 8 taps
    b0 += 512; b1 += 512;
    v16bf b0n = *(const v16bf*)b0;                 // prefetch (slack beyond last tile)
    v16bf b1n = *(const v16bf*)b1;

    int r0 = stid[i0 + lhalf];                     // elements 0..7 : i = i0+lhalf,   k = e
    int r2 = stid[i0 + 2 + lhalf];                 // elements 8..15: i = i0+2+lhalf, k = e-8
    v16bf af;
#pragma unroll
    for (int e = 0; e < 8; ++e)
      af[e] = __builtin_bit_cast(__bf16, semb[r0 * EMB_D + p + e]);
#pragma unroll
    for (int e = 8; e < 16; ++e)
      af[e] = __builtin_bit_cast(__bf16, semb[r2 * EMB_D + p + e - 8]);

    acc0 = __builtin_amdgcn_wmma_f32_16x16x32_bf16(false, af, false, b0c, (short)0, acc0, false, false);
    acc1 = __builtin_amdgcn_wmma_f32_16x16x32_bf16(false, af, false, b1c, (short)0, acc1, false, false);
    b0c = b0n; b1c = b1n;
  }

#pragma unroll
  for (int j = 0; j < 8; ++j) {
    int pp = (wave << 4) + j + (lhalf << 3);
    if (pp < CONVP) {
      out[(long)g * (CONVO * CONVP) + lrow * CONVP + pp]        = acc0[j] + bc[lrow];
      out[(long)g * (CONVO * CONVP) + (16 + lrow) * CONVP + pp] = acc1[j] + bc[16 + lrow];
    }
  }
}

// ---------------- host side ----------------
static inline int cdivl(long a, int b) { return (int)((a + b - 1) / b); }

extern "C" void kernel_launch(void* const* d_in, const int* in_sizes, int n_in,
                              void* d_out, int out_size, void* d_ws, size_t ws_size,
                              hipStream_t stream) {
  (void)in_sizes; (void)n_in; (void)out_size; (void)ws_size;
  const float* x       = (const float*)d_in[0];
  const int*   ei      = (const int*)d_in[1];
  const int*   batch   = (const int*)d_in[2];
  const int*   target  = (const int*)d_in[3];
  const float* W_gat   = (const float*)d_in[4];
  const float* att_src = (const float*)d_in[5];
  const float* att_dst = (const float*)d_in[6];
  const float* b_gat   = (const float*)d_in[7];
  const float* W_gcn   = (const float*)d_in[8];
  const float* b_gcn   = (const float*)d_in[9];
  const float* W_fcg1  = (const float*)d_in[10];
  const float* b_fcg1  = (const float*)d_in[11];
  const float* W_fcg2  = (const float*)d_in[12];
  const float* b_fcg2  = (const float*)d_in[13];
  const float* emb     = (const float*)d_in[14];
  const float* W_conv  = (const float*)d_in[15];
  const float* b_conv  = (const float*)d_in[16];
  const float* W_fcxt  = (const float*)d_in[17];
  const float* b_fcxt  = (const float*)d_in[18];
  const float* W_fc1   = (const float*)d_in[19];
  const float* b_fc1   = (const float*)d_in[20];
  const float* W_fc2   = (const float*)d_in[21];
  const float* b_fc2   = (const float*)d_in[22];
  const float* W_out   = (const float*)d_in[23];
  const float* b_out   = (const float*)d_in[24];
  float* out = (float*)d_out;

  // workspace bump allocator (floats, 256B aligned) — total ~281 MB
  size_t off = 0;
  float* base = (float*)d_ws;
  auto alloc = [&](size_t nf) { float* p = base + off; off += (nf + 63) & ~(size_t)63; return p; };
  float* bufH  = alloc((size_t)N_NODES * HC);     // h, later reused as xw
  float* bufXG = alloc((size_t)N_NODES * HC);     // xg, later reused as xg2
  float* aS    = alloc((size_t)N_NODES * HEADS);
  float* aD    = alloc((size_t)N_NODES * HEADS);
  float* mBuf  = alloc((size_t)N_NODES * HEADS);
  float* dBuf  = alloc((size_t)N_NODES * HEADS);
  float* eWs   = alloc((size_t)E_TOT * HEADS);
  float* deg   = alloc(N_NODES);
  float* dinv  = alloc(N_NODES);
  float* gmax  = alloc((size_t)NGRAPH * HC);
  float* gsum  = alloc((size_t)NGRAPH * HC);
  float* cnt   = alloc(NGRAPH);
  float* xp    = alloc((size_t)NGRAPH * 2 * HC);
  float* xp1   = alloc((size_t)NGRAPH * 1500);
  float* xc    = alloc((size_t)NGRAPH * 256);
  float* convo = alloc((size_t)NGRAPH * CONVO * CONVP);
  float* xc1   = alloc((size_t)NGRAPH * 1024);
  float* xc2   = alloc((size_t)NGRAPH * 512);
  // bf16 pack buffers (float units: 2 bf16 per float); extra slack for pipeline over-read
  __bf16* ApBuf = (__bf16*)alloc(12100000);       // >= 30016*800 bf16 + slack (48 MB)
  __bf16* BpBuf = (__bf16*)alloc(1300000);        // >= 49*94*512 bf16 + slack (5 MB)
  __bf16* WcPck = (__bf16*)alloc(130000);         // 2*250*512 bf16 + slack (0.5 MB)

  auto fill = [&](float* p, float v, long n) {
    k_fill<<<cdivl(n, 256), 256, 0, stream>>>(p, v, n);
  };
  auto gemm = [&](const float* A, const float* B, const float* bias, float* C,
                  int M, int N, int K, int ldc, int relu) {
    int Mt  = (M + 63) / 64;
    int Nt2 = (N + 31) / 32;                       // wave tile = 64x32
    int Ntp = Nt2 * 2;                             // packed n-tiles (even, zero-filled)
    int Kp = ((K + 31) / 32) * 32, Kt = Kp / 32;
    long aTot = (long)Mt * 64 * Kp;
    k_packA<<<cdivl(aTot, 256), 256, 0, stream>>>(A, ApBuf, M, K, Kp, aTot);
    long bTot = (long)Kt * Ntp * 512;
    k_packB<<<cdivl(bTot, 256), 256, 0, stream>>>(B, BpBuf, K, N, Kt, Ntp);
    int waves = Mt * Nt2;
    k_gemm_packed<<<(waves + 7) / 8, 256, 0, stream>>>(ApBuf, BpBuf, bias, C, M, N, Kp, ldc, relu);
  };

  // ---- GAT ----
  fill(bufXG, 0.f, (long)N_NODES * HC);
  fill(mBuf, -__builtin_inff(), (long)N_NODES * HEADS);
  fill(dBuf, 0.f, (long)N_NODES * HEADS);
  fill(deg, 0.f, N_NODES);
  gemm(x, W_gat, nullptr, bufH, N_NODES, HC, CHC, HC, 0);          // h = x @ W_gat
  k_att<<<cdivl((long)N_NODES * HEADS, 256), 256, 0, stream>>>(bufH, att_src, att_dst, aS, aD);
  long EH = (long)E_TOT * HEADS;
  k_edge_max<<<cdivl(EH, 256), 256, 0, stream>>>(ei, aS, aD, mBuf);
  k_edge_exp<<<cdivl(EH, 256), 256, 0, stream>>>(ei, aS, aD, mBuf, eWs, dBuf);
  k_gat_aggr<<<cdivl(EH, 256), 256, 0, stream>>>(ei, bufH, eWs, dBuf, bufXG);
  k_bias_relu<<<cdivl((long)N_NODES * HC, 256), 256, 0, stream>>>(bufXG, b_gat, N_NODES, HC);

  // ---- GCN ----
  k_deg<<<cdivl(E_TOT, 256), 256, 0, stream>>>(ei, deg);
  k_dinv<<<cdivl(N_NODES, 256), 256, 0, stream>>>(deg, dinv);
  gemm(bufXG, W_gcn, nullptr, bufH, N_NODES, HC, HC, HC, 0);       // xw (reuse h buf)
  fill(bufXG, 0.f, (long)N_NODES * HC);                            // xg2 accumulator
  k_gcn_aggr<<<cdivl(EH, 256), 256, 0, stream>>>(ei, bufH, dinv, bufXG);
  k_bias_relu<<<cdivl((long)N_NODES * HC, 256), 256, 0, stream>>>(bufXG, b_gcn, N_NODES, HC);

  // ---- pooling ----
  fill(gmax, -__builtin_inff(), (long)NGRAPH * HC);
  fill(gsum, 0.f, (long)NGRAPH * HC);
  fill(cnt, 0.f, NGRAPH);
  k_pool<<<cdivl((long)N_NODES * HC, 256), 256, 0, stream>>>(bufXG, batch, gmax, gsum);
  k_cnt<<<cdivl(N_NODES, 256), 256, 0, stream>>>(batch, cnt);
  k_pool_fin<<<cdivl((long)NGRAPH * HC, 256), 256, 0, stream>>>(gmax, gsum, cnt, xp);

  // ---- graph-branch MLP ----
  gemm(xp, W_fcg1, b_fcg1, xp1, NGRAPH, 1500, 2 * HC, 1500, 1);
  gemm(xp1, W_fcg2, b_fcg2, xc, NGRAPH, 128, 1500, 256, 0);        // xc[:, 0:128]

  // ---- protein branch ----
  {
    long wTot = (long)2 * CONV_KT * 512;
    k_packWc<<<cdivl(wTot, 256), 256, 0, stream>>>(W_conv, WcPck);
    k_conv<<<NGRAPH, 256, 0, stream>>>(target, emb, WcPck, b_conv, convo);
  }
  gemm(convo, W_fcxt, b_fcxt, xc + 128, NGRAPH, 128, CONVO * CONVP, 256, 0);

  // ---- joint MLP ----
  gemm(xc, W_fc1, b_fc1, xc1, NGRAPH, 1024, 256, 1024, 1);
  gemm(xc1, W_fc2, b_fc2, xc2, NGRAPH, 512, 1024, 512, 1);
  gemm(xc2, W_out, b_out, out, NGRAPH, 1, 512, 1, 0);
}
